// DecoderRNN_22892175688399
// MI455X (gfx1250) — matrix operand order repaired and checked by reading, not compiled, not run on hardware
//
#include <hip/hip_runtime.h>
#include <hip/hip_bf16.h>

typedef __attribute__((ext_vector_type(16))) __bf16 v16bf;
typedef __attribute__((ext_vector_type(8)))  float  v8f;
typedef __attribute__((ext_vector_type(4)))  int    v4i;

#define B_   32
#define T_   64
#define E_   512
#define H_   512
#define V_   10000
#define G4H  (4 * H_)   // 2048

#define LSTM_NBLK 4     // workgroups cooperating on the recurrence

// FC macro-tile: 128 (M) x 80 (N); 80*125 = 10000 and 128*16 = 2048 -> no edges
#define FC_MB   128
#define FC_NB   80
#define FC_KC   128     // K chunk staged in LDS per iteration (512/128 = 4 chunks)

#if __has_builtin(__builtin_amdgcn_global_load_async_to_lds_b128) && \
    __has_builtin(__builtin_amdgcn_s_wait_asynccnt)
#define HAVE_ASYNC_LDS 1
typedef __attribute__((address_space(1))) v4i* gptr_v4i;   // global (AS1)
typedef __attribute__((address_space(3))) v4i* lptr_v4i;   // LDS (AS3)
#endif

union FragBF {
    v16bf v;
    uint4 q[2];
};

// A-matrix (16x32 bf16) fragment from row-major [rows][ld] source.
// ISA 7.12.2: lanes 0-15 hold M=0-15; VGPR0-3 = K(kb..kb+7), VGPR4-7 = K(kb+16..kb+23),
// kb = (lane>=16 ? 8 : 0).
__device__ inline v16bf load_a_rm(const __bf16* __restrict__ src, int ld, int m0, int k0, int lane) {
    FragBF f;
    int row = m0 + (lane & 15);
    int kb  = (lane >> 4) << 3;
    const __bf16* p = src + row * ld + k0 + kb;
    f.q[0] = *reinterpret_cast<const uint4*>(p);
    f.q[1] = *reinterpret_cast<const uint4*>(p + 16);
    return f.v;
}

// B-matrix (32x16 bf16) fragment; W stored row-major [n][k] (B = W^T).
// column = lane&15; same K layout as A: elems 0-7 = K(kb..kb+7), 8-15 = K(kb+16..kb+23),
// kb = (lane>=16 ? 8 : 0).
__device__ inline v16bf load_b_wT(const __bf16* __restrict__ w, int ld, int n0, int k0, int lane) {
    FragBF f;
    int col = n0 + (lane & 15);
    int kb  = (lane >> 4) << 3;
    const __bf16* p = w + col * ld + k0 + kb;
    f.q[0] = *reinterpret_cast<const uint4*>(p);
    f.q[1] = *reinterpret_cast<const uint4*>(p + 16);
    return f.v;
}

__device__ inline float sigmoid_f(float x) { return 1.0f / (1.0f + __expf(-x)); }

#define WMMA_BF16(A, Bf, C) \
    __builtin_amdgcn_wmma_f32_16x16x32_bf16(false, (A), false, (Bf), (short)0, (C), false, false)

// ---------------------------------------------------------------------------
// Kernel 1: fp32->bf16 weights, xs = [features ; embed[captions]],
// zero the h exchange buffer and the step-barrier counters (every launch ->
// graph-replay safe).
// ---------------------------------------------------------------------------
__global__ void prep_kernel(const float* __restrict__ features,
                            const int*   __restrict__ captions,
                            const float* __restrict__ embed,
                            const float* __restrict__ wih_f,
                            const float* __restrict__ whh_f,
                            const float* __restrict__ fcw_f,
                            __bf16* __restrict__ wih_b,
                            __bf16* __restrict__ whh_b,
                            __bf16* __restrict__ fcw_b,
                            __bf16* __restrict__ xs,
                            __bf16* __restrict__ hg,
                            unsigned int* __restrict__ cnt) {
    const int stride = gridDim.x * blockDim.x;
    const int gid = blockIdx.x * blockDim.x + threadIdx.x;

    for (int i = gid; i < G4H * E_; i += stride) wih_b[i] = (__bf16)wih_f[i];
    for (int i = gid; i < G4H * H_; i += stride) whh_b[i] = (__bf16)whh_f[i];
    for (int i = gid; i < V_ * H_;  i += stride) fcw_b[i] = (__bf16)fcw_f[i];
    for (int i = gid; i < 2 * B_ * H_; i += stride) hg[i] = (__bf16)0.0f;
    for (int i = gid; i < T_; i += stride) cnt[i] = 0u;

    for (int i = gid; i < T_ * B_ * E_; i += stride) {
        int t = i / (B_ * E_);
        int r = i - t * (B_ * E_);
        int b = r >> 9;          // / E_
        int j = r & (E_ - 1);
        float v;
        if (t == 0) {
            v = features[b * E_ + j];
        } else {
            int tok = captions[b * T_ + (t - 1)];
            v = embed[(size_t)tok * E_ + j];
        }
        xs[i] = (__bf16)v;
    }
}

// ---------------------------------------------------------------------------
// Kernel 2: sequential LSTM split across LSTM_NBLK cooperating workgroups.
// Block p owns hidden columns [p*128, p*128+128); wave w owns one 16-column
// group and computes i/f/g/o for BOTH 16-row M-tiles (each weight fragment
// loaded once, used twice). h is exchanged through a double-buffered global
// bf16 buffer; steps are separated by a release/acquire atomic counter
// barrier (4 zero-LDS 256-thread blocks => co-residency guaranteed).
// K loops are clamped to unroll-2: enough to pipeline loads, small enough to
// avoid VGPR-MSB addressing / scratch spills seen with full unroll.
// ---------------------------------------------------------------------------
__global__ __launch_bounds__(256, 1)
void lstm_kernel(const __bf16* __restrict__ xs,     // [T][B][E] bf16
                 const __bf16* __restrict__ wih,    // [4H][E]   bf16
                 const __bf16* __restrict__ whh,    // [4H][H]   bf16
                 const float*  __restrict__ b_ih,   // [4H]
                 const float*  __restrict__ b_hh,   // [4H]
                 __bf16* __restrict__ hg,           // [2][B][H] exchange buffer
                 __bf16* __restrict__ hs,           // [T][B][H] output
                 unsigned int* __restrict__ cnt) {  // [T] barrier counters
    const int tid  = threadIdx.x;
    const int wave = tid >> 5;
    const int lane = tid & 31;
    const int ng   = blockIdx.x * 8 + wave;   // column group 0..31
    const int j0   = ng * 16;
    const int col  = j0 + (lane & 15);

    float creg[2][8];
#pragma unroll
    for (int mt = 0; mt < 2; ++mt)
#pragma unroll
        for (int r = 0; r < 8; ++r) creg[mt][r] = 0.0f;

    float biasv[4];
#pragma unroll
    for (int g = 0; g < 4; ++g) biasv[g] = b_ih[g * H_ + col] + b_hh[g * H_ + col];

    for (int t = 0; t < T_; ++t) {
        const __bf16* xt   = xs + (size_t)t * B_ * E_;
        const __bf16* hcur = hg + (size_t)(t & 1) * B_ * H_;
        __bf16*       hnxt = hg + (size_t)((t + 1) & 1) * B_ * H_;

        v8f acc[4][2];
#pragma unroll
        for (int g = 0; g < 4; ++g) { acc[g][0] = (v8f){}; acc[g][1] = (v8f){}; }

        // gates += x(t) @ W_ih^T : B fragment shared by both M-tiles
#pragma unroll 2
        for (int k0 = 0; k0 < E_; k0 += 32) {
            v16bf a0 = load_a_rm(xt, E_, 0,  k0, lane);
            v16bf a1 = load_a_rm(xt, E_, 16, k0, lane);
#pragma unroll
            for (int g = 0; g < 4; ++g) {
                v16bf b = load_b_wT(wih, E_, j0 + g * H_, k0, lane);
                acc[g][0] = WMMA_BF16(a0, b, acc[g][0]);
                acc[g][1] = WMMA_BF16(a1, b, acc[g][1]);
            }
        }
        // gates += h(t) @ W_hh^T
#pragma unroll 2
        for (int k0 = 0; k0 < H_; k0 += 32) {
            v16bf a0 = load_a_rm(hcur, H_, 0,  k0, lane);
            v16bf a1 = load_a_rm(hcur, H_, 16, k0, lane);
#pragma unroll
            for (int g = 0; g < 4; ++g) {
                v16bf b = load_b_wT(whh, H_, j0 + g * H_, k0, lane);
                acc[g][0] = WMMA_BF16(a0, b, acc[g][0]);
                acc[g][1] = WMMA_BF16(a1, b, acc[g][1]);
            }
        }

        // in-register cell update on the D-fragment layout:
        // VGPR r -> row mt*16 + r + 8*(lane>=16), column j0 + (lane&15)
#pragma unroll
        for (int mt = 0; mt < 2; ++mt) {
#pragma unroll
            for (int r = 0; r < 8; ++r) {
                int row = mt * 16 + r + ((lane >> 4) << 3);
                float ig = sigmoid_f(acc[0][mt][r] + biasv[0]);
                float fg = sigmoid_f(acc[1][mt][r] + biasv[1]);
                float gg = tanhf(acc[2][mt][r] + biasv[2]);
                float og = sigmoid_f(acc[3][mt][r] + biasv[3]);
                float c  = fg * creg[mt][r] + ig * gg;
                creg[mt][r] = c;
                float h = og * tanhf(c);
                __bf16 hb = (__bf16)h;
                hnxt[row * H_ + col] = hb;
                hs[(size_t)t * B_ * H_ + row * H_ + col] = hb;
            }
        }

        // inter-workgroup step barrier (release/acquire on cnt[t])
        __threadfence();
        __syncthreads();
        if (tid == 0) {
            __hip_atomic_fetch_add(&cnt[t], 1u, __ATOMIC_RELEASE, __HIP_MEMORY_SCOPE_AGENT);
            while (__hip_atomic_load(&cnt[t], __ATOMIC_ACQUIRE, __HIP_MEMORY_SCOPE_AGENT) <
                   (unsigned)LSTM_NBLK)
                __builtin_amdgcn_s_sleep(1);
        }
#if __has_builtin(__builtin_amdgcn_s_cluster_barrier)
        // CDNA5 cluster barrier: hardware-fast path if this grid is ever
        // dispatched as a workgroup cluster; architected NOP otherwise.
        __builtin_amdgcn_s_cluster_barrier();
#endif
        __syncthreads();
    }
}

// ---------------------------------------------------------------------------
// Kernel 3: logits[b][t][v] = hs[t][b] . fc_W[v] + fc_b[v]
// Block computes a 128x80 macro-tile. B panel chunks (80 x 128 bf16) are
// staged into double-buffered LDS (async global->LDS when available); each
// wave owns one M-tile and 5 N-tiles, so every A fragment is reused 5x.
// ---------------------------------------------------------------------------
__global__ __launch_bounds__(256, 2)
void fc_kernel(const __bf16* __restrict__ hs,    // [T*B][H] bf16
               const __bf16* __restrict__ fcw,   // [V][H]   bf16
               const float*  __restrict__ fcb,   // [V]
               float* __restrict__ out) {        // [B][T][V] f32
    __shared__ __bf16 Bl[2][FC_NB][FC_KC];       // 2 * 80 * 128 * 2B = 40 KB

    const int tid   = threadIdx.x;
    const int wave  = tid >> 5;
    const int lane  = tid & 31;
    const int m0    = blockIdx.y * FC_MB + wave * 16;
    const int nbase = blockIdx.x * FC_NB;

    // stage K-chunk `kc` of the B panel: 80 rows x 128 bf16 = 1280 uint4,
    // 256 threads x 5 transfers
    auto stage = [&](int kc) {
        __bf16* dst = &Bl[kc & 1][0][0];
#pragma unroll
        for (int p = 0; p < 5; ++p) {
            int f   = tid + p * 256;
            int row = f >> 4;           // 16 uint4 per 128-elem row
            int u   = f & 15;
            const __bf16* src = fcw + (size_t)(nbase + row) * H_ + kc * FC_KC + u * 8;
            __bf16* d = dst + row * FC_KC + u * 8;
#ifdef HAVE_ASYNC_LDS
            __builtin_amdgcn_global_load_async_to_lds_b128((gptr_v4i)(void*)src,
                                                           (lptr_v4i)(void*)d, 0, 0);
#else
            *reinterpret_cast<uint4*>(d) = *reinterpret_cast<const uint4*>(src);
#endif
        }
    };
    auto stage_wait = [&]() {
#ifdef HAVE_ASYNC_LDS
        __builtin_amdgcn_s_wait_asynccnt(0);
#endif
        __syncthreads();
    };

    v8f acc[5];
#pragma unroll
    for (int v = 0; v < 5; ++v) acc[v] = (v8f){};

    stage(0);
    stage_wait();
    for (int kc = 0; kc < H_ / FC_KC; ++kc) {
        if (kc + 1 < H_ / FC_KC) stage(kc + 1);
        const __bf16* bpan = &Bl[kc & 1][0][0];
#pragma unroll
        for (int k0 = 0; k0 < FC_KC; k0 += 32) {
            v16bf a = load_a_rm(hs, H_, m0, kc * FC_KC + k0, lane);
#pragma unroll
            for (int v = 0; v < 5; ++v) {
                v16bf b = load_b_wT(bpan, FC_KC, v * 16, k0, lane);
                acc[v] = WMMA_BF16(a, b, acc[v]);
            }
        }
        if (kc + 1 < H_ / FC_KC) stage_wait();
    }

    const int row0 = m0 + ((lane >> 4) << 3);
#pragma unroll
    for (int v = 0; v < 5; ++v) {
        int col  = nbase + v * 16 + (lane & 15);
        float bv = fcb[col];
#pragma unroll
        for (int r = 0; r < 8; ++r) {
            int gr = row0 + r;           // global row = t*B + b
            int t  = gr >> 5;
            int b  = gr & 31;
            out[((size_t)b * T_ + t) * V_ + col] = acc[v][r] + bv;
        }
    }
}

// ---------------------------------------------------------------------------
extern "C" void kernel_launch(void* const* d_in, const int* in_sizes, int n_in,
                              void* d_out, int out_size, void* d_ws, size_t ws_size,
                              hipStream_t stream) {
    (void)in_sizes; (void)n_in; (void)out_size; (void)ws_size;

    const float* features = (const float*)d_in[0];
    const int*   captions = (const int*)  d_in[1];
    // d_in[2] = forward_approach (unused by the reference math)
    const float* embed    = (const float*)d_in[3];
    const float* Wih      = (const float*)d_in[4];
    const float* Whh      = (const float*)d_in[5];
    const float* bih      = (const float*)d_in[6];
    const float* bhh      = (const float*)d_in[7];
    const float* fcW      = (const float*)d_in[8];
    const float* fcb      = (const float*)d_in[9];
    float* out = (float*)d_out;

    // workspace carve-out (~18.6 MB total)
    char*  ws  = (char*)d_ws;
    size_t off = 0;
    auto carve = [&](size_t bytes) {
        void* p = ws + off;
        off = (off + bytes + 255) & ~(size_t)255;
        return p;
    };
    __bf16* wih_b = (__bf16*)carve((size_t)G4H * E_ * sizeof(__bf16));      // 2 MB
    __bf16* whh_b = (__bf16*)carve((size_t)G4H * H_ * sizeof(__bf16));      // 2 MB
    __bf16* fcw_b = (__bf16*)carve((size_t)V_ * H_ * sizeof(__bf16));       // ~9.8 MB
    __bf16* xs    = (__bf16*)carve((size_t)T_ * B_ * E_ * sizeof(__bf16));  // 2 MB
    __bf16* hsb   = (__bf16*)carve((size_t)T_ * B_ * H_ * sizeof(__bf16));  // 2 MB
    __bf16* hg    = (__bf16*)carve((size_t)2 * B_ * H_ * sizeof(__bf16));   // 64 KB
    unsigned int* cnt = (unsigned int*)carve(T_ * sizeof(unsigned int));

    prep_kernel<<<1024, 256, 0, stream>>>(features, captions, embed, Wih, Whh, fcW,
                                          wih_b, whh_b, fcw_b, xs, hg, cnt);
    lstm_kernel<<<LSTM_NBLK, 256, 0, stream>>>(xs, wih_b, whh_b, bih, bhh, hg, hsb, cnt);
    fc_kernel<<<dim3(V_ / FC_NB, (T_ * B_) / FC_MB), 256, 0, stream>>>(hsb, fcw_b, fcb, out);
}
